// GPT_48945447305686
// MI455X (gfx1250) — matrix operand order repaired and checked by reading, not compiled, not run on hardware
//
#include <hip/hip_runtime.h>

// GPT-2 small forward on gfx1250: bf16 WMMA + async-to-LDS double-buffered GEMM.
// B=4, T=1024, D=768, H=12, HD=64, L=12, V=50257.

#define D_MODEL 768
#define NH      12
#define HDIM    64
#define SEQ     1024
#define NBATCH  4
#define NLAYER  12
#define NTOK    4096      // B*T
#define VOC     50257

typedef unsigned short u16;
typedef __attribute__((ext_vector_type(16))) __bf16 v16bf;
typedef __attribute__((ext_vector_type(8)))  float  v8f;
typedef int v4i __attribute__((vector_size(16)));   // matches builtin's V4i

union Frag16 { v16bf v; uint4 q[2]; };

__device__ __forceinline__ v8f wmma_bf16(v16bf a, v16bf b, v8f c) {
  // D = A(16x32) * B(32x16) + C, fp32 accumulate
  return __builtin_amdgcn_wmma_f32_16x16x32_bf16(false, a, false, b, (short)0, c,
                                                 false, false);
}

// A fragment (16x32 bf16). Row m = lane&15.
// elements 0..7  -> K = kbase + (lane>=16 ? 8:0) + 0..7
// elements 8..15 -> K = kbase + (lane>=16 ? 8:0) + 16..23
__device__ __forceinline__ v16bf load_a_frag(const u16* p0, int ld, int lane, int kbase) {
  int koff = kbase + ((lane & 16) ? 8 : 0);
  const u16* p = p0 + (size_t)(lane & 15) * ld + koff;
  Frag16 f;
  f.q[0] = *(const uint4*)(p);
  f.q[1] = *(const uint4*)(p + 16);
  return f.v;
}

// B fragment (32x16 bf16) from an N x K (row = output column) layout.
// Column n = lane&15 (memory row), elements 0..7 -> K = kbase + (lane>=16?8:0) + 0..7, 8..15 -> +16..23
__device__ __forceinline__ v16bf load_b_frag(const u16* p0, int ld, int lane, int kbase) {
  int koff = kbase + ((lane & 16) ? 8 : 0);
  const u16* p = p0 + (size_t)(lane & 15) * ld + koff;
  Frag16 f;
  f.q[0] = *(const uint4*)(p);
  f.q[1] = *(const uint4*)(p + 16);
  return f.v;
}

__device__ __forceinline__ u16 f2bf_bits(float x) {
  __bf16 h = (__bf16)x;
  return __builtin_bit_cast(u16, h);
}

// -------- CDNA5 async global->LDS copy (ASYNCcnt-tracked, bypasses VGPRs)

__device__ __forceinline__ void async_ld_b128(const u16* gsrc, u16* ldst) {
#if __has_builtin(__builtin_amdgcn_global_load_async_to_lds_b128)
  __builtin_amdgcn_global_load_async_to_lds_b128(
      (__attribute__((address_space(1))) v4i*)gsrc,
      (__attribute__((address_space(3))) v4i*)ldst, 0, 0);
#else
  // flat LDS address low 32 bits == LDS byte offset (aperture truncation)
  unsigned loff = (unsigned)(unsigned long long)(size_t)ldst;
  asm volatile("global_load_async_to_lds_b128 %0, %1, off"
               :: "v"(loff), "v"((unsigned long long)(size_t)gsrc)
               : "memory");
#endif
}

template <int N>
__device__ __forceinline__ void wait_async() {
#if __has_builtin(__builtin_amdgcn_s_wait_asynccnt)
  __builtin_amdgcn_s_wait_asynccnt(N);
#else
  asm volatile("s_wait_asynccnt %0" :: "n"(N) : "memory");
#endif
}

// ---------------------------------------------------------------- converts

__global__ __launch_bounds__(256) void cast_bf16_kernel(const float* __restrict__ s,
                                                        __bf16* __restrict__ d, int n) {
  int i = blockIdx.x * 256 + threadIdx.x;
  if (i < n) d[i] = (__bf16)s[i];
}

// src (L, K, N) fp32 -> dst (L, N, K) bf16
__global__ __launch_bounds__(256) void transpose_cast_kernel(const float* __restrict__ s,
                                                             __bf16* __restrict__ d,
                                                             int L, int K, int N) {
  int i = blockIdx.x * 256 + threadIdx.x;
  int total = L * K * N;
  if (i >= total) return;
  int n = i % N;
  int k = (i / N) % K;
  int l = i / (N * K);
  d[((size_t)l * N + n) * K + k] = (__bf16)s[i];
}

// ---------------------------------------------------------------- embed / LN

__global__ __launch_bounds__(256) void embed_kernel(const int* __restrict__ idx,
                                                    const float* __restrict__ wte,
                                                    const float* __restrict__ wpe,
                                                    float* __restrict__ x) {
  int row = blockIdx.x;                 // 0..NTOK-1
  int tok = idx[row];
  int t = row & (SEQ - 1);
  const float* we = wte + (size_t)tok * D_MODEL;
  const float* pe = wpe + (size_t)t * D_MODEL;
  float* xr = x + (size_t)row * D_MODEL;
  for (int c = threadIdx.x; c < D_MODEL; c += 256) xr[c] = we[c] + pe[c];
}

__global__ __launch_bounds__(256) void layernorm_kernel(const float* __restrict__ x,
                                                        const float* __restrict__ w,
                                                        const float* __restrict__ b,
                                                        __bf16* __restrict__ out) {
  __shared__ float red0[8], red1[8];
  int row = blockIdx.x;
  const float* xr = x + (size_t)row * D_MODEL;
  float s = 0.f, s2 = 0.f;
  for (int c = threadIdx.x; c < D_MODEL; c += 256) {
    float v = xr[c];
    s += v; s2 += v * v;
  }
#pragma unroll
  for (int m = 1; m < 32; m <<= 1) {
    s  += __shfl_xor(s,  m, 32);
    s2 += __shfl_xor(s2, m, 32);
  }
  int wv = threadIdx.x >> 5, lane = threadIdx.x & 31;
  if (lane == 0) { red0[wv] = s; red1[wv] = s2; }
  __syncthreads();
  float ts = 0.f, ts2 = 0.f;
#pragma unroll
  for (int i = 0; i < 8; ++i) { ts += red0[i]; ts2 += red1[i]; }
  float mu  = ts * (1.0f / D_MODEL);
  float var = ts2 * (1.0f / D_MODEL) - mu * mu;
  float rinv = rsqrtf(var + 1e-5f);
  __bf16* orow = out + (size_t)row * D_MODEL;
  for (int c = threadIdx.x; c < D_MODEL; c += 256)
    orow[c] = (__bf16)((xr[c] - mu) * rinv * w[c] + b[c]);
}

// ---------------------------------------------------------------- WMMA GEMM
// C(M,N) = A(M,K)bf16 * BT(N,K)bf16 (+bias) with templated epilogue.
// Block: 256 threads (8 waves). Tile 128x128, BK=64, double-buffered LDS
// filled by async global->LDS copies (8 async instructions per wave per tile,
// uniform so s_wait_asynccnt bookkeeping is exact).
// wave (wm in 0..1, wn in 0..3): 64x32 sub-tile = 4x2 accumulators,
// 16 WMMAs per K-tile between barriers.

#define BM  128
#define BN  128
#define BK  64
#define LDT 72   // padded row stride (shorts); 144B rows, 16B aligned

#define EPI_GELU   1
#define EPI_RES    2
#define EPI_QKV    3
#define EPI_LOGITS 4

template <int EPI>
__global__ __launch_bounds__(256) void gemm_bf16_kernel(
    const __bf16* __restrict__ A, const __bf16* __restrict__ BT,
    const float* __restrict__ bias, const float* __restrict__ resid,
    float* __restrict__ outf, __bf16* __restrict__ outh,
    __bf16* __restrict__ q_buf, __bf16* __restrict__ k_buf,
    __bf16* __restrict__ vt_buf, int M, int N, int K) {
  __shared__ u16 a_s[2][BM * LDT];
  __shared__ u16 b_s[2][BN * LDT];

  int tid = threadIdx.x;
  int lane = tid & 31;
  int wv = tid >> 5;
  int wm = wv >> 2, wn = wv & 3;
  int m0 = blockIdx.y * BM;
  int n0 = blockIdx.x * BN;

  v8f acc[4][2];
#pragma unroll
  for (int i = 0; i < 4; ++i)
#pragma unroll
    for (int j = 0; j < 2; ++j)
#pragma unroll
      for (int r = 0; r < 8; ++r) acc[i][j][r] = 0.f;

  int ldrow = tid >> 1;             // 0..127
  int ldc0  = (tid & 1) * 32;       // 0 or 32 (shorts)
  const u16* Ag = (const u16*)A;
  const u16* Bg = (const u16*)BT;

  // clamp B row for the logits N-edge: keeps async issue count uniform;
  // out-of-range columns are discarded by the store guard in the epilogue.
  int gnb = n0 + ldrow;
  if (gnb > N - 1) gnb = N - 1;

  auto stage = [&](int buf, int kb) {
    const u16* pa = Ag + (size_t)(m0 + ldrow) * K + kb + ldc0;
    u16* da = &a_s[buf][ldrow * LDT + ldc0];
#pragma unroll
    for (int c = 0; c < 4; ++c) async_ld_b128(pa + c * 8, da + c * 8);
    const u16* pb = Bg + (size_t)gnb * K + kb + ldc0;
    u16* db = &b_s[buf][ldrow * LDT + ldc0];
#pragma unroll
    for (int c = 0; c < 4; ++c) async_ld_b128(pb + c * 8, db + c * 8);
  };

  stage(0, 0);
  int nk = K / BK;
  for (int t = 0; t < nk; ++t) {
    int cur = t & 1;
    if (t + 1 < nk) {
      stage(cur ^ 1, (t + 1) * BK);  // prefetch next tile into other buffer
      wait_async<8>();               // current tile (first 8 in-order) done
    } else {
      wait_async<0>();
    }
    __syncthreads();

    const u16* as = a_s[cur];
    const u16* bs = b_s[cur];
#pragma unroll
    for (int kk = 0; kk < BK; kk += 32) {
      v16bf af[4], bfr[2];
#pragma unroll
      for (int i = 0; i < 4; ++i)
        af[i] = load_a_frag(as + (wm * 64 + i * 16) * LDT, LDT, lane, kk);
#pragma unroll
      for (int j = 0; j < 2; ++j)
        bfr[j] = load_b_frag(bs + (wn * 32 + j * 16) * LDT, LDT, lane, kk);
#pragma unroll
      for (int i = 0; i < 4; ++i)
#pragma unroll
        for (int j = 0; j < 2; ++j)
          acc[i][j] = wmma_bf16(af[i], bfr[j], acc[i][j]);
    }
    __syncthreads();  // all waves done reading before buffer is refilled
  }

  // epilogue: element r -> row = base+r (+8 for hi half-wave), col = lane&15
  int cn = lane & 15;
  int rb = (lane & 16) ? 8 : 0;
#pragma unroll
  for (int i = 0; i < 4; ++i) {
#pragma unroll
    for (int j = 0; j < 2; ++j) {
      int gn = n0 + wn * 32 + j * 16 + cn;
      if (EPI == EPI_LOGITS && gn >= N) continue;
      float bv = 0.f;
      if (EPI != EPI_LOGITS) bv = bias[gn];
      int gm_base = m0 + wm * 64 + i * 16 + rb;
#pragma unroll
      for (int r = 0; r < 8; ++r) {
        int gm = gm_base + r;
        float v = acc[i][j][r] + bv;
        if (EPI == EPI_GELU) {
          float g = 0.5f * v * (1.0f + erff(v * 0.70710678118f));
          outh[(size_t)gm * N + gn] = (__bf16)g;
        } else if (EPI == EPI_RES) {
          outf[(size_t)gm * N + gn] = resid[(size_t)gm * N + gn] + v;
        } else if (EPI == EPI_QKV) {
          int bidx = gm >> 10, t2 = gm & (SEQ - 1);
          if (gn < D_MODEL) {
            int h = gn >> 6, d = gn & 63;
            q_buf[(((size_t)bidx * NH + h) * SEQ + t2) * HDIM + d] = (__bf16)v;
          } else if (gn < 2 * D_MODEL) {
            int c = gn - D_MODEL, h = c >> 6, d = c & 63;
            k_buf[(((size_t)bidx * NH + h) * SEQ + t2) * HDIM + d] = (__bf16)v;
          } else {
            int c = gn - 2 * D_MODEL, h = c >> 6, d = c & 63;
            vt_buf[(((size_t)bidx * NH + h) * HDIM + d) * SEQ + t2] = (__bf16)v;
          }
        } else {  // logits
          outf[(size_t)gm * N + gn] = v;
        }
      }
    }
  }
}

// ---------------------------------------------------------------- attention
// Flash-style causal attention. Block = 256 thr = 8 waves; each wave owns 16
// queries of one (b,h). grid = (T/128, H, B). No block barriers (waves have
// causally different trip counts); LDS regions are per-wave; LDS ops are
// in-order within a wave so the C->A reshape through LDS is hazard-free.

__global__ __launch_bounds__(256) void attn_kernel(const __bf16* __restrict__ qg,
                                                   const __bf16* __restrict__ kg,
                                                   const __bf16* __restrict__ vtg,
                                                   __bf16* __restrict__ og) {
  __shared__ u16 p_lds[8][16][32];

  int tid = threadIdx.x, lane = tid & 31, wv = tid >> 5;
  int b = blockIdx.z, h = blockIdx.y;
  int q0 = blockIdx.x * 128 + wv * 16;  // query tile base within sequence

  const u16* qh = (const u16*)(qg + (((size_t)b * NH + h) * SEQ) * HDIM);
  const u16* kh = (const u16*)(kg + (((size_t)b * NH + h) * SEQ) * HDIM);
  const u16* vh = (const u16*)(vtg + (((size_t)b * NH + h) * HDIM) * SEQ);

  // Q fragments (held in registers across the whole K loop)
  v16bf aq0 = load_a_frag(qh + (size_t)q0 * HDIM, HDIM, lane, 0);
  v16bf aq1 = load_a_frag(qh + (size_t)q0 * HDIM, HDIM, lane, 32);

  v8f o[4];
#pragma unroll
  for (int jt = 0; jt < 4; ++jt)
#pragma unroll
    for (int r = 0; r < 8; ++r) o[jt][r] = 0.f;

  float mrow[8], lrow[8];
#pragma unroll
  for (int r = 0; r < 8; ++r) { mrow[r] = -1e30f; lrow[r] = 0.f; }

  const float scale = 0.125f;  // 1/sqrt(64)
  int colk = lane & 15;
  int rbase = q0 + ((lane & 16) ? 8 : 0);
  int mrow_lds = ((lane & 16) ? 8 : 0);

  for (int kk0 = 0; kk0 < q0 + 16; kk0 += 32) {
    v8f s0, s1;
#pragma unroll
    for (int r = 0; r < 8; ++r) { s0[r] = 0.f; s1[r] = 0.f; }

    // scores: S = Q(16x64) @ K^T -> two 16x16 tiles, K-dim split 32+32
    v16bf bk;
    bk = load_b_frag(kh + (size_t)kk0 * HDIM, HDIM, lane, 0);
    s0 = wmma_bf16(aq0, bk, s0);
    bk = load_b_frag(kh + (size_t)kk0 * HDIM, HDIM, lane, 32);
    s0 = wmma_bf16(aq1, bk, s0);
    bk = load_b_frag(kh + (size_t)(kk0 + 16) * HDIM, HDIM, lane, 0);
    s1 = wmma_bf16(aq0, bk, s1);
    bk = load_b_frag(kh + (size_t)(kk0 + 16) * HDIM, HDIM, lane, 32);
    s1 = wmma_bf16(aq1, bk, s1);

    int key0 = kk0 + colk;
    int key1 = kk0 + 16 + colk;

#pragma unroll
    for (int r = 0; r < 8; ++r) {
      int qrow = rbase + r;
      float v0 = s0[r] * scale;
      float v1 = s1[r] * scale;
      if (key0 > qrow) v0 = -1e30f;
      if (key1 > qrow) v1 = -1e30f;
      // row reduction across the 16 lanes holding this row's columns
      float tmax = fmaxf(v0, v1);
#pragma unroll
      for (int m = 1; m < 16; m <<= 1) tmax = fmaxf(tmax, __shfl_xor(tmax, m, 32));
      float mnew = fmaxf(mrow[r], tmax);
      float alpha = __expf(mrow[r] - mnew);
      float p0 = __expf(v0 - mnew);
      float p1 = __expf(v1 - mnew);
      float ps = p0 + p1;
#pragma unroll
      for (int m = 1; m < 16; m <<= 1) ps += __shfl_xor(ps, m, 32);
      lrow[r] = lrow[r] * alpha + ps;
      mrow[r] = mnew;
#pragma unroll
      for (int jt = 0; jt < 4; ++jt) o[jt][r] *= alpha;
      // stage P (bf16) into per-wave LDS tile, C-layout -> row-major 16x32
      p_lds[wv][mrow_lds + r][colk]      = f2bf_bits(p0);
      p_lds[wv][mrow_lds + r][16 + colk] = f2bf_bits(p1);
    }

    // reload P as an A fragment (16x32); same-wave LDS ops are in-order
    v16bf ap = load_a_frag(&p_lds[wv][0][0], 32, lane, 0);
#pragma unroll
    for (int jt = 0; jt < 4; ++jt) {
      v16bf bv = load_b_frag(vh + (size_t)(jt * 16) * SEQ, SEQ, lane, kk0);
      o[jt] = wmma_bf16(ap, bv, o[jt]);
    }
  }

  // normalize and write: out (B*T, 768) bf16, col = h*64 + jt*16 + lane&15
#pragma unroll
  for (int r = 0; r < 8; ++r) {
    float inv = 1.0f / lrow[r];
    int tok = b * SEQ + rbase + r;
#pragma unroll
    for (int jt = 0; jt < 4; ++jt)
      og[(size_t)tok * D_MODEL + h * HDIM + jt * 16 + colk] = (__bf16)(o[jt][r] * inv);
  }
}

// ---------------------------------------------------------------- launch

extern "C" void kernel_launch(void* const* d_in, const int* in_sizes, int n_in,
                              void* d_out, int out_size, void* d_ws, size_t ws_size,
                              hipStream_t stream) {
  (void)in_sizes; (void)n_in; (void)out_size; (void)ws_size;

  const int*   idx   = (const int*)d_in[0];
  const float* wte   = (const float*)d_in[1];
  const float* wpe   = (const float*)d_in[2];
  const float* ln1w  = (const float*)d_in[3];
  const float* ln1b  = (const float*)d_in[4];
  const float* qkvw  = (const float*)d_in[5];
  const float* qkvb  = (const float*)d_in[6];
  const float* outw  = (const float*)d_in[7];
  const float* outb  = (const float*)d_in[8];
  const float* ln2w  = (const float*)d_in[9];
  const float* ln2b  = (const float*)d_in[10];
  const float* fc1w  = (const float*)d_in[11];
  const float* fc1b  = (const float*)d_in[12];
  const float* fc2w  = (const float*)d_in[13];
  const float* fc2b  = (const float*)d_in[14];
  const float* lnfw  = (const float*)d_in[15];
  const float* lnfb  = (const float*)d_in[16];
  float* logits = (float*)d_out;

  char* p = (char*)d_ws;
  auto take = [&](size_t bytes) {
    char* r = p;
    p += (bytes + 255) & ~(size_t)255;
    return r;
  };
  __bf16* wte_h = (__bf16*)take((size_t)VOC * D_MODEL * 2);
  __bf16* qkvT  = (__bf16*)take((size_t)NLAYER * 3 * D_MODEL * D_MODEL * 2);
  __bf16* outT  = (__bf16*)take((size_t)NLAYER * D_MODEL * D_MODEL * 2);
  __bf16* fc1T  = (__bf16*)take((size_t)NLAYER * 4 * D_MODEL * D_MODEL * 2);
  __bf16* fc2T  = (__bf16*)take((size_t)NLAYER * 4 * D_MODEL * D_MODEL * 2);
  float*  x     = (float*)take((size_t)NTOK * D_MODEL * 4);
  __bf16* hbuf  = (__bf16*)take((size_t)NTOK * D_MODEL * 2);
  __bf16* act1  = (__bf16*)take((size_t)NTOK * 4 * D_MODEL * 2);
  __bf16* q_buf = (__bf16*)take((size_t)NTOK * D_MODEL * 2);
  __bf16* k_buf = (__bf16*)take((size_t)NTOK * D_MODEL * 2);
  __bf16* vt_buf= (__bf16*)take((size_t)NTOK * D_MODEL * 2);
  __bf16* o_buf = (__bf16*)take((size_t)NTOK * D_MODEL * 2);

  // one-time (per launch) weight conversions, bf16 transposed (N,K)
  {
    int n = VOC * D_MODEL;
    cast_bf16_kernel<<<(n + 255) / 256, 256, 0, stream>>>(wte, wte_h, n);
  }
  {
    int n = NLAYER * D_MODEL * 3 * D_MODEL;
    transpose_cast_kernel<<<(n + 255) / 256, 256, 0, stream>>>(qkvw, qkvT, NLAYER, D_MODEL, 3 * D_MODEL);
  }
  {
    int n = NLAYER * D_MODEL * D_MODEL;
    transpose_cast_kernel<<<(n + 255) / 256, 256, 0, stream>>>(outw, outT, NLAYER, D_MODEL, D_MODEL);
  }
  {
    int n = NLAYER * D_MODEL * 4 * D_MODEL;
    transpose_cast_kernel<<<(n + 255) / 256, 256, 0, stream>>>(fc1w, fc1T, NLAYER, D_MODEL, 4 * D_MODEL);
    transpose_cast_kernel<<<(n + 255) / 256, 256, 0, stream>>>(fc2w, fc2T, NLAYER, 4 * D_MODEL, D_MODEL);
  }

  embed_kernel<<<NTOK, 256, 0, stream>>>(idx, wte, wpe, x);

  for (int l = 0; l < NLAYER; ++l) {
    layernorm_kernel<<<NTOK, 256, 0, stream>>>(x, ln1w + l * D_MODEL, ln1b + l * D_MODEL, hbuf);

    gemm_bf16_kernel<EPI_QKV><<<dim3(3 * D_MODEL / BN, NTOK / BM), 256, 0, stream>>>(
        hbuf, qkvT + (size_t)l * 3 * D_MODEL * D_MODEL, qkvb + (size_t)l * 3 * D_MODEL,
        nullptr, nullptr, nullptr, q_buf, k_buf, vt_buf, NTOK, 3 * D_MODEL, D_MODEL);

    attn_kernel<<<dim3(SEQ / 128, NH, NBATCH), 256, 0, stream>>>(q_buf, k_buf, vt_buf, o_buf);

    gemm_bf16_kernel<EPI_RES><<<dim3(D_MODEL / BN, NTOK / BM), 256, 0, stream>>>(
        o_buf, outT + (size_t)l * D_MODEL * D_MODEL, outb + (size_t)l * D_MODEL,
        x, x, nullptr, nullptr, nullptr, nullptr, NTOK, D_MODEL, D_MODEL);

    layernorm_kernel<<<NTOK, 256, 0, stream>>>(x, ln2w + l * D_MODEL, ln2b + l * D_MODEL, hbuf);

    gemm_bf16_kernel<EPI_GELU><<<dim3(4 * D_MODEL / BN, NTOK / BM), 256, 0, stream>>>(
        hbuf, fc1T + (size_t)l * 4 * D_MODEL * D_MODEL, fc1b + (size_t)l * 4 * D_MODEL,
        nullptr, nullptr, act1, nullptr, nullptr, nullptr, NTOK, 4 * D_MODEL, D_MODEL);

    gemm_bf16_kernel<EPI_RES><<<dim3(D_MODEL / BN, NTOK / BM), 256, 0, stream>>>(
        act1, fc2T + (size_t)l * 4 * D_MODEL * D_MODEL, fc2b + (size_t)l * D_MODEL,
        x, x, nullptr, nullptr, nullptr, nullptr, NTOK, D_MODEL, 4 * D_MODEL);
  }

  layernorm_kernel<<<NTOK, 256, 0, stream>>>(x, lnfw, lnfb, hbuf);

  // logits = h @ wte^T  : wte (V, D) is already (N, K) for the NT GEMM
  gemm_bf16_kernel<EPI_LOGITS><<<dim3((VOC + BN - 1) / BN, NTOK / BM), 256, 0, stream>>>(
      hbuf, wte_h, nullptr, nullptr, logits, nullptr, nullptr, nullptr, nullptr,
      NTOK, VOC, D_MODEL);
}
